// Attention_5102421148295
// MI455X (gfx1250) — compile-verified, matrix operand order checked by reading, not run on hardware
//
#include <hip/hip_runtime.h>
#include <hip/hip_bf16.h>

typedef __bf16 bf16;
typedef __attribute__((ext_vector_type(16))) __bf16 v16bf;
typedef __attribute__((ext_vector_type(8)))  __bf16 v8bf;
typedef __attribute__((ext_vector_type(8)))  float  v8f;

#define WMMA_BF16(a, b, c) \
    __builtin_amdgcn_wmma_f32_16x16x32_bf16(false, (a), false, (b), (short)0, (c), false, false)

union Frag {
    v16bf v;
    v8bf  h[2];
    bf16  e[16];
};

// Load a 16x32 bf16 A/B fragment (wave32 layout) from an M-/N-major staged
// tile: lane L owns row rn (already includes lane&15); its 16 elements are two
// contiguous 8-element chunks at K offsets {kb, 16+kb}, kb = (L>=16)*8.
__device__ __forceinline__ v16bf load_frag(const bf16* base, int rn, int ld, int lane) {
    const int kb = (lane >> 4) * 8;
    Frag f;
    const bf16* p = base + rn * ld + kb;
    f.h[0] = *(const v8bf*)(p);
    f.h[1] = *(const v8bf*)(p + 16);
    return f.v;
}

// ---------------------------------------------------------------------------
// Kernel 1: qkv = x(131072x512,f32) @ w_qkv(512x1536,f32) -> bf16, q pre-scaled
// Block tile 128x128, 256 threads (8 waves), wave tile 32x64 (2x4 WMMA tiles).
// Coalesced bf16 writeback via wave-private LDS transpose (b128 stores).
// ---------------------------------------------------------------------------
__global__ __launch_bounds__(256)
void qkv_gemm_kernel(const float* __restrict__ x, const float* __restrict__ w,
                     bf16* __restrict__ qkv) {
    __shared__ bf16 ldsA[128 * 40];       // [m][k], pitch 40 (80B rows, 16B aligned)
    __shared__ bf16 ldsBT[128 * 40];      // [n][k] (transposed at stage time)
    __shared__ bf16 ldsC[8 * 32 * 72];    // per-wave 32x64 C staging, pitch 72

    const int t    = threadIdx.x;
    const int lane = t & 31;
    const int wv   = t >> 5;
    const int n0   = blockIdx.x * 128;
    const long m0  = (long)blockIdx.y * 128;
    const int wm   = (wv & 3) * 32;       // wave m offset in block
    const int wn   = (wv >> 2) * 64;      // wave n offset in block

    v8f acc[2][4] = {};

    const int arow = t >> 1, ac16 = (t & 1) * 16;        // A staging coords
    const int brow = t >> 3, bc16 = (t & 7) * 16;        // B staging coords

    for (int k0 = 0; k0 < 512; k0 += 32) {
        { // stage A: 128x32 fp32 -> bf16 (16 elems/thread)
            const float* src = x + (m0 + arow) * 512 + k0 + ac16;
            Frag fa;
            #pragma unroll
            for (int e = 0; e < 16; ++e) fa.e[e] = (bf16)src[e];
            v8bf* dst = (v8bf*)(ldsA + arow * 40 + ac16);
            dst[0] = fa.h[0];
            dst[1] = fa.h[1];
            if (k0 + 32 < 512) __builtin_prefetch(src + 32, 0, 1);
        }
        { // stage B transposed: 32x128 fp32 -> ldsBT[n][k] (16 elems/thread)
            const float* src = w + (size_t)(k0 + brow) * 1536 + n0 + bc16;
            #pragma unroll
            for (int e = 0; e < 16; ++e)
                ldsBT[(bc16 + e) * 40 + brow] = (bf16)src[e];
            if (k0 + 32 < 512) __builtin_prefetch(src + 32 * 1536, 0, 1);
        }
        __syncthreads();

        v16bf afrag[2], bfrag[4];
        #pragma unroll
        for (int mi = 0; mi < 2; ++mi)
            afrag[mi] = load_frag(ldsA, wm + mi * 16 + (lane & 15), 40, lane);
        #pragma unroll
        for (int ni = 0; ni < 4; ++ni)
            bfrag[ni] = load_frag(ldsBT, wn + ni * 16 + (lane & 15), 40, lane);
        #pragma unroll
        for (int mi = 0; mi < 2; ++mi)
            #pragma unroll
            for (int ni = 0; ni < 4; ++ni)
                acc[mi][ni] = WMMA_BF16(afrag[mi], bfrag[ni], acc[mi][ni]);
        __syncthreads();
    }

    // q section scale is block-uniform: whole 128-col block is q iff n0 < 512
    const float s = (n0 < 512) ? 0.125f : 1.0f;

    // transpose C fragments through wave-private LDS, then coalesced b128 stores
    bf16* cw = ldsC + wv * (32 * 72);
    #pragma unroll
    for (int mi = 0; mi < 2; ++mi)
        #pragma unroll
        for (int ni = 0; ni < 4; ++ni)
            #pragma unroll
            for (int r = 0; r < 8; ++r)
                cw[(mi * 16 + r + 8 * (lane >> 4)) * 72 + ni * 16 + (lane & 15)] =
                    (bf16)(acc[mi][ni][r] * s);

    const int lr = lane >> 3;           // 0..3
    const int lc = (lane & 7) * 8;      // 0..56 (16B chunks)
    #pragma unroll
    for (int p = 0; p < 8; ++p) {
        const int row = p * 4 + lr;
        const v8bf cv = *(const v8bf*)(cw + row * 72 + lc);
        const long grow = m0 + wm + row;
        *(v8bf*)(qkv + grow * 1536 + n0 + wn + lc) = cv;
    }
}

// ---------------------------------------------------------------------------
// Kernel 2: per (b, head): S = Q^T K (64x64, reduce over 4096), softmax,
//           out = V attn^T. One block (256 threads / 8 waves) per (b,head).
// ---------------------------------------------------------------------------
__global__ __launch_bounds__(256)
void attn_kernel(const bf16* __restrict__ qkv, bf16* __restrict__ outw) {
    __shared__ bf16  ldsQT[64 * 40];     // [i][d]  (transposed Q chunk)
    __shared__ bf16  ldsKT[64 * 40];     // [j][d]  (transposed K chunk)
    __shared__ float Sbuf[64 * 68];      // scores fp32
    __shared__ bf16  attnLds[64 * 72];   // attn[i][j] bf16 (== N-major B staging)
    __shared__ bf16  ldsO[8 * 16 * 72];  // per-wave 16x64 out staging

    const int t    = threadIdx.x;
    const int lane = t & 31;
    const int wv   = t >> 5;
    const int b    = blockIdx.x >> 3;
    const int head = blockIdx.x & 7;
    const long nbase = (long)b * 4096;

    const int ti  = wv >> 1;          // i tile 0..3
    const int tj0 = (wv & 1) * 2;     // j tiles {tj0, tj0+1}

    // ---- Phase 1: S = Q^T K, 128 k-steps of 32 over d ----
    v8f sacc[2] = {};
    const int r = t >> 3;             // d row 0..31
    const int c = (t & 7) * 8;        // channel base 0..56
    for (int d0 = 0; d0 < 4096; d0 += 32) {
        const bf16* qsrc = qkv + (nbase + d0 + r) * 1536 + head * 64 + c;
        const v8bf qv = *(const v8bf*)(qsrc);
        const v8bf kv = *(const v8bf*)(qsrc + 512);
        if (d0 + 32 < 4096) {
            __builtin_prefetch(qsrc + 32 * 1536, 0, 1);
            __builtin_prefetch(qsrc + 32 * 1536 + 512, 0, 1);
        }
        #pragma unroll
        for (int e = 0; e < 8; ++e) {
            ldsQT[(c + e) * 40 + r] = qv[e];
            ldsKT[(c + e) * 40 + r] = kv[e];
        }
        __syncthreads();

        const v16bf aq = load_frag(ldsQT, ti * 16 + (lane & 15), 40, lane);
        #pragma unroll
        for (int j = 0; j < 2; ++j) {
            const v16bf bk = load_frag(ldsKT, (tj0 + j) * 16 + (lane & 15), 40, lane);
            sacc[j] = WMMA_BF16(aq, bk, sacc[j]);
        }
        __syncthreads();
    }

    // ---- Phase 2: softmax over j ----
    #pragma unroll
    for (int j = 0; j < 2; ++j)
        #pragma unroll
        for (int rr = 0; rr < 8; ++rr)
            Sbuf[(ti * 16 + rr + 8 * (lane >> 4)) * 68 + (tj0 + j) * 16 + (lane & 15)] =
                sacc[j][rr];
    __syncthreads();

    if (t < 64) {
        float m = -1e30f;
        for (int j = 0; j < 64; ++j) m = fmaxf(m, Sbuf[t * 68 + j]);
        float sum = 0.0f;
        for (int j = 0; j < 64; ++j) sum += __expf(Sbuf[t * 68 + j] - m);
        const float inv = 1.0f / sum;
        for (int j = 0; j < 64; ++j)
            attnLds[t * 72 + j] = (bf16)(__expf(Sbuf[t * 68 + j] - m) * inv);
    }
    __syncthreads();

    // ---- Phase 3: out[d, i] = sum_j V[d, j] * attn[i, j] ----
    // B fragments (attn^T) live in registers for the whole phase.
    v16bf bfr[4][2];
    #pragma unroll
    for (int it = 0; it < 4; ++it)
        #pragma unroll
        for (int kc = 0; kc < 2; ++kc)
            bfr[it][kc] = load_frag(attnLds + kc * 32, it * 16 + (lane & 15), 72, lane);

    const bf16* vbase = qkv + 1024 + head * 64;   // V section, this head
    const int kb = (lane >> 4) * 8;
    bf16* ow = ldsO + wv * (16 * 72);
    const int lr = lane >> 3;                     // 0..3
    const int lc = (lane & 7) * 8;                // 0..56

    for (int dt = 0; dt < 32; ++dt) {             // 512 rows per wave
        const int d0 = wv * 512 + dt * 16;
        const long row = nbase + d0 + (lane & 15);
        v8f oacc[4] = {};
        #pragma unroll
        for (int kc = 0; kc < 2; ++kc) {
            Frag fa;   // A = V tile, loaded straight from global (16B chunks)
            const bf16* src = vbase + row * 1536 + kc * 32 + kb;
            fa.h[0] = *(const v8bf*)(src);
            fa.h[1] = *(const v8bf*)(src + 16);
            #pragma unroll
            for (int it = 0; it < 4; ++it)
                oacc[it] = WMMA_BF16(fa.v, bfr[it][kc], oacc[it]);
        }
        // stage 16x64 tile in wave-private LDS, then coalesced b128 stores
        #pragma unroll
        for (int it = 0; it < 4; ++it)
            #pragma unroll
            for (int rr = 0; rr < 8; ++rr)
                ow[(rr + 8 * (lane >> 4)) * 72 + it * 16 + (lane & 15)] = (bf16)oacc[it][rr];
        #pragma unroll
        for (int p = 0; p < 4; ++p) {
            const int orow = p * 4 + lr;
            const v8bf cv = *(const v8bf*)(ow + orow * 72 + lc);
            *(v8bf*)(outw + (nbase + d0 + orow) * 512 + head * 64 + lc) = cv;
        }
    }
}

// ---------------------------------------------------------------------------
// Kernel 3: d_out = outw(131072x512,bf16) @ w_out(512x512,f32) + b_out  (f32)
// Block tile 128x128, wave tile 32x64. fp32 output stores are coalesced as-is.
// ---------------------------------------------------------------------------
__global__ __launch_bounds__(256)
void out_gemm_kernel(const bf16* __restrict__ a, const float* __restrict__ w,
                     const float* __restrict__ bias, float* __restrict__ out) {
    __shared__ bf16 ldsA[128 * 40];
    __shared__ bf16 ldsBT[128 * 40];

    const int t    = threadIdx.x;
    const int lane = t & 31;
    const int wv   = t >> 5;
    const int n0   = blockIdx.x * 128;
    const long m0  = (long)blockIdx.y * 128;
    const int wm   = (wv & 3) * 32;
    const int wn   = (wv >> 2) * 64;

    v8f acc[2][4] = {};

    const int arow = t >> 1, ac16 = (t & 1) * 16;
    const int brow = t >> 3, bc16 = (t & 7) * 16;

    for (int k0 = 0; k0 < 512; k0 += 32) {
        { // stage A: already bf16, straight 32B copy per thread
            const v8bf* src = (const v8bf*)(a + (m0 + arow) * 512 + k0 + ac16);
            v8bf* dst = (v8bf*)(ldsA + arow * 40 + ac16);
            dst[0] = src[0];
            dst[1] = src[1];
            if (k0 + 32 < 512) __builtin_prefetch(src + 2, 0, 1);
        }
        { // stage B transposed: 32x128 fp32 -> bf16
            const float* src = w + (size_t)(k0 + brow) * 512 + n0 + bc16;
            #pragma unroll
            for (int e = 0; e < 16; ++e)
                ldsBT[(bc16 + e) * 40 + brow] = (bf16)src[e];
            if (k0 + 32 < 512) __builtin_prefetch(src + 32 * 512, 0, 1);
        }
        __syncthreads();

        v16bf afrag[2], bfrag[4];
        #pragma unroll
        for (int mi = 0; mi < 2; ++mi)
            afrag[mi] = load_frag(ldsA, wm + mi * 16 + (lane & 15), 40, lane);
        #pragma unroll
        for (int ni = 0; ni < 4; ++ni)
            bfrag[ni] = load_frag(ldsBT, wn + ni * 16 + (lane & 15), 40, lane);
        #pragma unroll
        for (int mi = 0; mi < 2; ++mi)
            #pragma unroll
            for (int ni = 0; ni < 4; ++ni)
                acc[mi][ni] = WMMA_BF16(afrag[mi], bfrag[ni], acc[mi][ni]);
        __syncthreads();
    }

    #pragma unroll
    for (int mi = 0; mi < 2; ++mi)
        #pragma unroll
        for (int ni = 0; ni < 4; ++ni) {
            const int gcol = n0 + wn + ni * 16 + (lane & 15);
            const float bv = bias[gcol];
            #pragma unroll
            for (int rr = 0; rr < 8; ++rr) {
                const long grow = m0 + wm + mi * 16 + rr + 8 * (lane >> 4);
                out[grow * 512 + gcol] = acc[mi][ni][rr] + bv;
            }
        }
}

// ---------------------------------------------------------------------------
extern "C" void kernel_launch(void* const* d_in, const int* in_sizes, int n_in,
                              void* d_out, int out_size, void* d_ws, size_t ws_size,
                              hipStream_t stream) {
    (void)in_sizes; (void)n_in; (void)out_size; (void)ws_size;

    const float* x     = (const float*)d_in[0];   // (32,64,64,512)
    const float* w_qkv = (const float*)d_in[1];   // (512,1536)
    const float* w_out = (const float*)d_in[2];   // (512,512)
    const float* b_out = (const float*)d_in[3];   // (512,)

    bf16* qkv  = (bf16*)d_ws;                          // 131072 x 1536 bf16 (~403MB)
    bf16* outw = qkv + (size_t)131072 * 1536;          // 131072 x 512  bf16 (~134MB)

    // M=131072, K=512, N=1536
    qkv_gemm_kernel<<<dim3(1536 / 128, 131072 / 128), 256, 0, stream>>>(x, w_qkv, qkv);
    // 32 batches x 8 heads
    attn_kernel<<<dim3(256), 256, 0, stream>>>(qkv, outw);
    // M=131072, K=512, N=512
    out_gemm_kernel<<<dim3(512 / 128, 131072 / 128), 256, 0, stream>>>(outw, w_out, b_out,
                                                                       (float*)d_out);
}
